// Attention_10952166605242
// MI455X (gfx1250) — matrix operand order repaired and checked by reading, not compiled, not run on hardware
//
#include <hip/hip_runtime.h>
#include <hip/hip_bf16.h>

// ---------------------------------------------------------------------------
// MHA forward for MI455X (gfx1250): bf16 WMMA, flash softmax, async LDS staging
// B=2, N=2048, C=1024, H=16, DH=64.
// ---------------------------------------------------------------------------

typedef __bf16 bf16_t;
typedef __attribute__((ext_vector_type(16))) __bf16 v16bf;
typedef __attribute__((ext_vector_type(8)))  __bf16 v8bf;
typedef __attribute__((ext_vector_type(8)))  float  v8f;
typedef __attribute__((ext_vector_type(4)))  int    v4i;

union V16 { v16bf v; v8bf h[2]; };

#define NEG_HUGE (-3.4028234e38f)   // ~ -(finfo.max - 1000): reproduces torch clamp semantics
#define ATT_SCALE 0.125f            // 64^-0.5

#define WMMA_BF16(A, Bv, Cacc) \
  __builtin_amdgcn_wmma_f32_16x16x32_bf16(false, (A), false, (Bv), (short)0, (Cacc), false, false)

#define HAVE_ASYNC_LDS __has_builtin(__builtin_amdgcn_global_load_async_to_lds_b128)

// builtin signature (from hipcc diagnostic): (v4i AS1*, v4i AS3*, imm, imm)
#define AS1V4(p) ((__attribute__((address_space(1))) v4i*)(p))
#define AS3V4(p) ((__attribute__((address_space(3))) v4i*)(p))

#if __has_builtin(__builtin_amdgcn_s_wait_asynccnt)
#define WAIT_ASYNC(n) __builtin_amdgcn_s_wait_asynccnt(n)
#else
#define WAIT_ASYNC(n) asm volatile("s_wait_asynccnt " #n ::: "memory")
#endif

// ------------------------------- conversion --------------------------------

__global__ void k_cvt(const float* __restrict__ src, bf16_t* __restrict__ dst, int n) {
  int i = blockIdx.x * blockDim.x + threadIdx.x;
  if (i < n) dst[i] = (bf16_t)src[i];
}

// dst[R][C] = src[C][R] (src row-major [C rows][R cols]); bf16 transpose-convert
__global__ void k_cvt_T(const float* __restrict__ src, bf16_t* __restrict__ dst, int R, int C) {
  int i = blockIdx.x * blockDim.x + threadIdx.x;
  if (i < R * C) {
    int r = i / C, c = i % C;
    dst[i] = (bf16_t)src[(size_t)c * R + r];
  }
}

// ------------------------------ qkv GEMM -----------------------------------
// qkv = xb[4096,1024] @ wT[3072,1024]^T.  One wave -> 16x64 tile.
// All 8 waves of a block share the same n-tile -> identical B stream reuses WGP$.
__global__ __launch_bounds__(256) void k_gemm_qkv(
    const bf16_t* __restrict__ xb, const bf16_t* __restrict__ wT,
    bf16_t* __restrict__ Qm, bf16_t* __restrict__ Km, bf16_t* __restrict__ VTm) {
  int ln   = threadIdx.x & 31;
  int wv   = threadIdx.x >> 5;
  int nt   = blockIdx.x % 48;               // shared across the 8 waves
  int mt   = (blockIdx.x / 48) * 8 + wv;    // per-wave m tile
  int m0   = mt * 16;
  int c    = ln & 15;
  int half = ln >> 4;
  int kb   = half * 8;    // unused; A loads also use hb16 (contiguous K)
  int hb16 = half * 16;   // B-operand K base (contiguous 16 K per lane)

  const bf16_t* arow = xb + (size_t)(m0 + c) * 1024;
  v8f acc[4] = {};
  for (int k0 = 0; k0 < 1024; k0 += 32) {
    V16 a;
    a.h[0] = *(const v8bf*)(arow + k0 + hb16);
    a.h[1] = *(const v8bf*)(arow + k0 + hb16 + 8);
#pragma unroll
    for (int t = 0; t < 4; ++t) {
      v16bf bv = *(const v16bf*)(wT + (size_t)(nt * 64 + t * 16 + c) * 1024 + k0 + hb16);
      acc[t] = WMMA_BF16(a.v, bv, acc[t]);
    }
  }
  int qkv = nt >> 4;            // 0=q 1=k 2=v (64-wide n-tile never straddles)
  int h   = nt & 15;
  int b   = m0 >> 11;
  int bh  = b * 16 + h;
  int nb  = m0 & 2047;
#pragma unroll
  for (int t = 0; t < 4; ++t) {
#pragma unroll
    for (int r = 0; r < 8; ++r) {
      int n  = nb + r + half * 8;           // C-layout row ownership
      int dh = t * 16 + c;                  // C-layout col ownership
      bf16_t val = (bf16_t)acc[t][r];
      if (qkv == 0)      Qm[((size_t)bh * 2048 + n) * 64 + dh] = val;
      else if (qkv == 1) Km[((size_t)bh * 2048 + n) * 64 + dh] = val;
      else               VTm[((size_t)bh * 64 + dh) * 2048 + n] = val;  // transposed for P@V
    }
  }
}

// ---------------------------- flash attention ------------------------------
// Block = 8 waves, all on the same (b,h); each wave owns 16 query rows.
// K/V tiles (32 keys) staged cooperatively into double-buffered LDS with
// global_load_async_to_lds_b128 (ASYNCcnt) when available, overlapping WMMA.
__global__ __launch_bounds__(256) void k_attn(
    const bf16_t* __restrict__ Qm, const bf16_t* __restrict__ Km,
    const bf16_t* __restrict__ VTm, const int* __restrict__ maskp,
    bf16_t* __restrict__ AO) {
  __shared__ __align__(32) bf16_t Kls[2][32 * 64];   // 2 x 4KB key tiles (row-major keys x dh)
  __shared__ __align__(32) bf16_t Vls[2][64 * 32];   // 2 x 4KB V^T tiles (dh x keys)
  __shared__ __align__(32) bf16_t pshared[8][16 * 32];
  int tid  = threadIdx.x;
  int ln   = tid & 31;
  int wv   = tid >> 5;
  int w    = blockIdx.x * 8 + wv;
  int bh   = blockIdx.x >> 4;          // uniform across the block
  int q0   = (w & 127) * 16;
  int b    = bh >> 4;
  int c    = ln & 15;
  int half = ln >> 4;
  int kb   = half * 8;
  int hb16 = half * 16;
  bf16_t* pl = pshared[wv];

  const bf16_t* Kbase = Km  + (size_t)bh * 2048 * 64;
  const bf16_t* Vbase = VTm + (size_t)bh * 64 * 2048;

  // cooperative staging: 256 threads x one b128 each for K and for V
  int krow = tid >> 3, kcol = (tid & 7) * 8;   // K tile [32][64]
  int vrow = tid >> 2, vcol = (tid & 3) * 8;   // V tile [64][32]
  auto stage = [&](int k0, int buf) {
    const bf16_t* gk = Kbase + (size_t)(k0 + krow) * 64 + kcol;
    const bf16_t* gv = Vbase + (size_t)vrow * 2048 + k0 + vcol;
    bf16_t* lk = &Kls[buf][krow * 64 + kcol];
    bf16_t* lv = &Vls[buf][vrow * 32 + vcol];
#if HAVE_ASYNC_LDS
    __builtin_amdgcn_global_load_async_to_lds_b128(AS1V4(gk), AS3V4(lk), 0, 0);
    __builtin_amdgcn_global_load_async_to_lds_b128(AS1V4(gv), AS3V4(lv), 0, 0);
#else
    *(uint4*)lk = *(const uint4*)gk;
    *(uint4*)lv = *(const uint4*)gv;
#endif
  };

  // Q A-operands for the two dh-chunks (live across the whole key loop)
  const bf16_t* qrow = Qm + ((size_t)bh * 2048 + q0 + c) * 64;
  V16 qa0, qa1;
  qa0.h[0] = *(const v8bf*)(qrow + hb16);
  qa0.h[1] = *(const v8bf*)(qrow + hb16 + 8);
  qa1.h[0] = *(const v8bf*)(qrow + 32 + hb16);
  qa1.h[1] = *(const v8bf*)(qrow + 32 + hb16 + 8);

  float m_run[8], l_run[8];
#pragma unroll
  for (int r = 0; r < 8; ++r) { m_run[r] = NEG_HUGE; l_run[r] = 0.f; }
  v8f o[4] = {};

  const int* mrow = maskp + b * 2048;

  stage(0, 0);
  for (int k0 = 0, it = 0; k0 < 2048; k0 += 32, ++it) {
    int buf = it & 1;
    if (k0 + 32 < 2048) {
      stage(k0 + 32, buf ^ 1);      // prefetch next tile while we compute this one
#if HAVE_ASYNC_LDS
      WAIT_ASYNC(2);                // tile `it` fully landed; only next tile outstanding
#endif
    } else {
#if HAVE_ASYNC_LDS
      WAIT_ASYNC(0);
#endif
    }
    __syncthreads();                // all waves' slices of buf are visible

    // ---- S = Q @ K^T from LDS (two 16x16 key sub-tiles) ----
    const bf16_t* kr0 = &Kls[buf][c * 64 + hb16];
    const bf16_t* kr1 = kr0 + 16 * 64;
    v8f s0 = {}, s1 = {};
    s0 = WMMA_BF16(qa0.v, *(const v16bf*)(kr0),      s0);
    s0 = WMMA_BF16(qa1.v, *(const v16bf*)(kr0 + 32), s0);
    s1 = WMMA_BF16(qa0.v, *(const v16bf*)(kr1),      s1);
    s1 = WMMA_BF16(qa1.v, *(const v16bf*)(kr1 + 32), s1);

    // ---- mask + scale ----
    int mk0 = mrow[k0 + c];
    int mk1 = mrow[k0 + 16 + c];
    float s0v[8], s1v[8], rmax[8];
#pragma unroll
    for (int r = 0; r < 8; ++r) {
      s0v[r] = mk0 ? s0[r] * ATT_SCALE : NEG_HUGE;
      s1v[r] = mk1 ? s1[r] * ATT_SCALE : NEG_HUGE;
      rmax[r] = fmaxf(s0v[r], s1v[r]);
    }
    // row max across the 16 lanes of each half-wave (C-layout row striping)
#pragma unroll
    for (int d = 1; d < 16; d <<= 1)
#pragma unroll
      for (int r = 0; r < 8; ++r)
        rmax[r] = fmaxf(rmax[r], __shfl_xor(rmax[r], d, 32));

    float p0[8], p1[8], rsum[8], scl[8];
#pragma unroll
    for (int r = 0; r < 8; ++r) {
      float mn = fmaxf(m_run[r], rmax[r]);
      scl[r]  = __expf(m_run[r] - mn);
      m_run[r] = mn;
      p0[r] = __expf(s0v[r] - mn);
      p1[r] = __expf(s1v[r] - mn);
      rsum[r] = p0[r] + p1[r];
    }
#pragma unroll
    for (int d = 1; d < 16; d <<= 1)
#pragma unroll
      for (int r = 0; r < 8; ++r)
        rsum[r] += __shfl_xor(rsum[r], d, 32);
#pragma unroll
    for (int r = 0; r < 8; ++r) l_run[r] = l_run[r] * scl[r] + rsum[r];
#pragma unroll
    for (int t = 0; t < 4; ++t)
#pragma unroll
      for (int r = 0; r < 8; ++r) o[t][r] *= scl[r];

    // ---- C-layout P -> LDS -> A-operand layout (wave-private; DS in-order) ----
#pragma unroll
    for (int r = 0; r < 8; ++r) {
      pl[(r + half * 8) * 32 + c]      = (bf16_t)p0[r];
      pl[(r + half * 8) * 32 + 16 + c] = (bf16_t)p1[r];
    }
    V16 pa;
    const bf16_t* prow = pl + c * 32;
    pa.h[0] = *(const v8bf*)(prow + hb16);
    pa.h[1] = *(const v8bf*)(prow + hb16 + 8);

    // ---- O += P @ V from LDS ----
#pragma unroll
    for (int t = 0; t < 4; ++t) {
      v16bf vb = *(const v16bf*)(&Vls[buf][(t * 16 + c) * 32 + hb16]);
      o[t] = WMMA_BF16(pa.v, vb, o[t]);
    }
    __syncthreads();                // buf may be overwritten two stages from now
  }

  // normalize + store attention output [B*N, H*DH] in bf16 for the proj GEMM
  int hcol = (bh & 15) * 64;
#pragma unroll
  for (int t = 0; t < 4; ++t) {
#pragma unroll
    for (int r = 0; r < 8; ++r) {
      int n = q0 + r + half * 8;
      AO[((size_t)b * 2048 + n) * 1024 + hcol + t * 16 + c] = (bf16_t)(o[t][r] / l_run[r]);
    }
  }
}

// ----------------------------- output proj ---------------------------------
__global__ __launch_bounds__(256) void k_gemm_proj(
    const bf16_t* __restrict__ AO, const bf16_t* __restrict__ wT,
    const float* __restrict__ bias, float* __restrict__ out) {
  int ln   = threadIdx.x & 31;
  int wv   = threadIdx.x >> 5;
  int nt   = blockIdx.x % 16;               // shared across the 8 waves
  int mt   = (blockIdx.x / 16) * 8 + wv;
  int m0   = mt * 16;
  int n0   = nt * 64;
  int c    = ln & 15;
  int half = ln >> 4;
  int kb   = half * 8;
  int hb16 = half * 16;

  const bf16_t* arow = AO + (size_t)(m0 + c) * 1024;
  v8f acc[4] = {};
  for (int k0 = 0; k0 < 1024; k0 += 32) {
    V16 a;
    a.h[0] = *(const v8bf*)(arow + k0 + hb16);
    a.h[1] = *(const v8bf*)(arow + k0 + hb16 + 8);
#pragma unroll
    for (int t = 0; t < 4; ++t) {
      v16bf bv = *(const v16bf*)(wT + (size_t)(n0 + t * 16 + c) * 1024 + k0 + hb16);
      acc[t] = WMMA_BF16(a.v, bv, acc[t]);
    }
  }
#pragma unroll
  for (int t = 0; t < 4; ++t) {
    float bv = bias[n0 + t * 16 + c];
#pragma unroll
    for (int r = 0; r < 8; ++r)
      out[(size_t)(m0 + r + half * 8) * 1024 + n0 + t * 16 + c] = acc[t][r] + bv;
  }
}

// ------------------------------- launcher ----------------------------------

extern "C" void kernel_launch(void* const* d_in, const int* in_sizes, int n_in,
                              void* d_out, int out_size, void* d_ws, size_t ws_size,
                              hipStream_t stream) {
  const float* x      = (const float*)d_in[0];   // [2,2048,1024]
  const int*   mask   = (const int*)d_in[1];     // [2,2048]
  const float* w_qkv  = (const float*)d_in[2];   // [1024,3072]
  const float* w_proj = (const float*)d_in[3];   // [1024,1024]
  const float* b_proj = (const float*)d_in[4];   // [1024]
  float* out = (float*)d_out;                    // [2,2048,1024] fp32

  char* ws = (char*)d_ws;
  bf16_t* xb    = (bf16_t*)(ws + 0);             //  8.0 MB  x in bf16
  bf16_t* wqkvT = (bf16_t*)(ws + 8388608);       //  6.0 MB  w_qkv^T  [3072][1024]
  bf16_t* wpT   = (bf16_t*)(ws + 14680064);      //  2.0 MB  w_proj^T [1024][1024]
  bf16_t* Qm    = (bf16_t*)(ws + 16777216);      //  8.0 MB  [32][2048][64]
  bf16_t* Km    = (bf16_t*)(ws + 25165824);      //  8.0 MB  [32][2048][64]
  bf16_t* VTm   = (bf16_t*)(ws + 33554432);      //  8.0 MB  [32][64][2048]
  bf16_t* AO    = (bf16_t*)(ws + 41943040);      //  8.0 MB  [4096][1024]

  k_cvt  <<<(4194304 + 255) / 256, 256, 0, stream>>>(x, xb, 4194304);
  k_cvt_T<<<(3145728 + 255) / 256, 256, 0, stream>>>(w_qkv, wqkvT, 3072, 1024);
  k_cvt_T<<<(1048576 + 255) / 256, 256, 0, stream>>>(w_proj, wpT, 1024, 1024);

  k_gemm_qkv <<<1536, 256, 0, stream>>>(xb, wqkvT, Qm, Km, VTm);   // 12288 waves
  k_attn     <<< 512, 256, 0, stream>>>(Qm, Km, VTm, mask, AO);    //  4096 waves
  k_gemm_proj<<< 512, 256, 0, stream>>>(AO, wpT, b_proj, out);     //  4096 waves
}
